// SAModule_72945724555784
// MI455X (gfx1250) — matrix operand order repaired and checked by reading, not compiled, not run on hardware
//
#include <hip/hip_runtime.h>
#include <hip/hip_bf16.h>
#include <float.h>

// ---------------------------------------------------------------------------
// PointNet++ SA module for MI455X (gfx1250, wave32, WMMA + TDM).
//   B=8 clouds, N=4096 pts, C=3 feats, ratio=0.5 -> M=2048, radius=0.2, K=64
// Pipeline: fps_kernel -> ball_kernel (TDM-staged LDS) -> pointconv_kernel
// (f16 WMMA MLP).
// ---------------------------------------------------------------------------

#define BB 8
#define NN 4096
#define MM 2048
#define KK 64
#define R2 0.04f

typedef _Float16 half16 __attribute__((ext_vector_type(16)));
typedef _Float16 half8  __attribute__((ext_vector_type(8)));
typedef float    float8 __attribute__((ext_vector_type(8)));
typedef unsigned int u32x4 __attribute__((ext_vector_type(4)));
typedef unsigned int u32x8 __attribute__((ext_vector_type(8)));

#define F8Z (float8){0.f,0.f,0.f,0.f,0.f,0.f,0.f,0.f}

// ---------------------------------------------------------------------------
// Kernel 1: farthest point sampling. One block per batch, 1024 threads.
// Each thread keeps its 4 points and running min-distances in REGISTERS;
// per-iteration cost is a few VALU ops + a two-level wave32 argmax.
// Tie-break: smallest index (matches jnp.argmax first-max semantics).
// ---------------------------------------------------------------------------
__global__ __launch_bounds__(1024) void fps_kernel(const float* __restrict__ pos,
                                                   float* __restrict__ newPos,
                                                   int* __restrict__ outIdx) {
  const int b = blockIdx.x;
  const int tid = threadIdx.x;
  const float* pb = pos + (size_t)b * NN * 3;

  __shared__ int   sidx[MM];
  __shared__ float swv[32];
  __shared__ int   swj[32];
  __shared__ float sLast[3];

  // registers: 4 points per thread, strided
  float px[4], py[4], pz[4], dd[4];
#pragma unroll
  for (int r = 0; r < 4; r++) {
    int j = tid + r * 1024;
    px[r] = pb[3 * j + 0];
    py[r] = pb[3 * j + 1];
    pz[r] = pb[3 * j + 2];
    dd[r] = 1e10f;
  }
  if (tid == 0) {
    sidx[0] = 0;
    sLast[0] = pb[0]; sLast[1] = pb[1]; sLast[2] = pb[2];
  }
  __syncthreads();

  for (int i = 1; i < MM; i++) {
    const float lx = sLast[0], ly = sLast[1], lz = sLast[2];
    float bv = -1.f;
    int bj = 0x7fffffff;
#pragma unroll
    for (int r = 0; r < 4; r++) {
      int j = tid + r * 1024;
      float dx = px[r] - lx, dy = py[r] - ly, dz = pz[r] - lz;
      float d = dx * dx + dy * dy + dz * dz;
      float nd = fminf(dd[r], d);
      dd[r] = nd;
      if (nd > bv || (nd == bv && j < bj)) { bv = nd; bj = j; }
    }
    // wave32 argmax (tie -> smaller index)
#pragma unroll
    for (int off = 16; off > 0; off >>= 1) {
      float ov = __shfl_xor(bv, off, 32);
      int   oj = __shfl_xor(bj, off, 32);
      if (ov > bv || (ov == bv && oj < bj)) { bv = ov; bj = oj; }
    }
    if ((tid & 31) == 0) { swv[tid >> 5] = bv; swj[tid >> 5] = bj; }
    __syncthreads();
    if (tid < 32) {
      bv = swv[tid]; bj = swj[tid];
#pragma unroll
      for (int off = 16; off > 0; off >>= 1) {
        float ov = __shfl_xor(bv, off, 32);
        int   oj = __shfl_xor(bj, off, 32);
        if (ov > bv || (ov == bv && oj < bj)) { bv = ov; bj = oj; }
      }
      if (tid == 0) {
        sidx[i] = bj;
        sLast[0] = pb[3 * bj + 0];
        sLast[1] = pb[3 * bj + 1];
        sLast[2] = pb[3 * bj + 2];
      }
    }
    __syncthreads();
  }

  for (int i = tid; i < MM; i += 1024) {
    int j = sidx[i];
    outIdx[(size_t)b * MM + i] = j;
    newPos[((size_t)b * MM + i) * 3 + 0] = pb[3 * j + 0];
    newPos[((size_t)b * MM + i) * 3 + 1] = pb[3 * j + 1];
    newPos[((size_t)b * MM + i) * 3 + 2] = pb[3 * j + 2];
  }
}

// ---------------------------------------------------------------------------
// Kernel 2: exact K=64 nearest neighbors per centroid (ordered ascending by
// (d2, idx) — identical ordering to top_k(-d2) with index tie-break).
// One wave per centroid; the batch's 48KB pos block is staged into LDS by the
// Tensor Data Mover (tensor_load_to_lds, TENSORcnt-tracked) instead of
// burning vector lanes; then 64 selection passes, each a lexicographic min
// over 128 points/lane + wave reduction.
// ---------------------------------------------------------------------------
__global__ __launch_bounds__(256) void ball_kernel(const float* __restrict__ pos,
                                                   const float* __restrict__ newPos,
                                                   int* __restrict__ nbr,
                                                   float* __restrict__ nbrD2) {
  __shared__ __align__(16) float spos[NN * 3];
  const int tid  = threadIdx.x;
  const int lane = tid & 31;
  const int w    = tid >> 5;
  const int blk  = blockIdx.x;
  const int b    = blk >> 8;                    // 256 blocks per batch
  const int m    = ((blk & 255) << 3) + w;      // centroid within batch
  const float* pb = pos + (size_t)b * NN * 3;

  // ---- TDM: DMA 12288 f32 (48KB) global -> LDS, 1-row tile descriptor ----
  if (tid < 32) {   // wave 0 issues (TDM ignores EXEC; per-wave instruction)
    const unsigned lds_addr = (unsigned)(uintptr_t)(&spos[0]); // low 32b of
                                                               // flat-LDS ptr
    const unsigned long long ga = (unsigned long long)(uintptr_t)pb;
    const unsigned elems = NN * 3;              // 12288 4-byte elements
    u32x4 g0;
    g0[0] = 1u;                                  // count=1, user descriptor
    g0[1] = lds_addr;                            // D#.lds_addr (bytes)
    g0[2] = (unsigned)(ga & 0xffffffffu);        // global_addr[31:0]
    g0[3] = (unsigned)((ga >> 32) & 0x1ffffffu)  // global_addr[56:32]
            | (2u << 30);                        // type=2 ("image")
    u32x8 g1;
    g1[0] = 2u << 16;                            // data_size=2 -> 4 bytes
    g1[1] = (elems & 0xffffu) << 16;             // tensor_dim0[15:0]
    g1[2] = (elems >> 16) | (1u << 16);          // tensor_dim0[31:16], dim1=1
    g1[3] = (elems & 0xffffu) << 16;             // tile_dim0 = 12288
    g1[4] = 0u;                                  // tile_dim1/2 unused
    g1[5] = elems;                               // tensor_dim0_stride[31:0]
    g1[6] = 0u;
    g1[7] = 0u;
    asm volatile("tensor_load_to_lds %0, %1" :: "s"(g0), "s"(g1) : "memory");
    __builtin_amdgcn_s_wait_tensorcnt(0);
  }
  __syncthreads();

  const size_t gm = (size_t)b * MM + m;
  const float qx = newPos[gm * 3 + 0];
  const float qy = newPos[gm * 3 + 1];
  const float qz = newPos[gm * 3 + 2];

  float prevD = -1.f;
  int   prevJ = -1;
  for (int k = 0; k < KK; k++) {
    float bv = FLT_MAX;
    int   bj = 0x7fffffff;
    for (int j = lane; j < NN; j += 32) {
      float dx = spos[3 * j + 0] - qx;
      float dy = spos[3 * j + 1] - qy;
      float dz = spos[3 * j + 2] - qz;
      float d = dx * dx + dy * dy + dz * dz;
      bool cand = (d > prevD) || (d == prevD && j > prevJ);
      if (cand && (d < bv || (d == bv && j < bj))) { bv = d; bj = j; }
    }
#pragma unroll
    for (int off = 16; off > 0; off >>= 1) {
      float ov = __shfl_xor(bv, off, 32);
      int   oj = __shfl_xor(bj, off, 32);
      if (ov < bv || (ov == bv && oj < bj)) { bv = ov; bj = oj; }
    }
    prevD = bv;
    prevJ = bj;
    if (lane == 0) {
      nbr[gm * KK + k]   = bj;
      nbrD2[gm * KK + k] = bv;
    }
  }
}

// ---------------------------------------------------------------------------
// WMMA fragment loaders (contiguous k = 16h + s order, same for A and B, wave32).
// A 16x32 f16: lane<16 holds row (lane), K k0..k0+15;
//              lane>=16 holds row (lane-16), K k0+16..k0+31.
// B 32x16 f16 from TRANSPOSED weights Wt[n][k] so each lane's 16 halfs are
// contiguous: lane<16 -> col lane, K k0..k0+15; lane>=16 -> col, K k0+16..+31.
// ---------------------------------------------------------------------------
__device__ __forceinline__ half16 load_afrag(const _Float16* base, int ld,
                                             int row0, int k0, int lane) {
  const int ln = lane & 15;
  const int hi = lane >> 4;
  const _Float16* p = base + (row0 + ln) * ld + k0 + hi * 16;
  union { half16 v; half8 h[2]; } u;
  u.h[0] = *(const half8*)(p);
  u.h[1] = *(const half8*)(p + 8);
  return u.v;
}

__device__ __forceinline__ half16 load_bfrag(const _Float16* base, int ld,
                                             int col0, int k0, int lane) {
  const int ln = lane & 15;
  const int hi = lane >> 4;
  const _Float16* p = base + (col0 + ln) * ld + k0 + hi * 16;
  return *(const half16*)(p);
}

__device__ __forceinline__ float8 wmma_f16(half16 a, half16 b, float8 c) {
  return __builtin_amdgcn_wmma_f32_16x16x32_f16(false, a, false, b,
                                                (short)0, c, false, false);
}

// ---------------------------------------------------------------------------
// Kernel 3: PointConv MLP via f16 WMMA. One block (8 waves) per centroid.
//   A  = [64 x 6->32pad]  (x_j ++ (pos_j - q))
//   L1 = relu(A @ W1 + b1)   [64x64]   16 WMMAs
//   L2 = relu(L1 @ W2 + b2)  [64x64]   32 WMMAs
//   L3 = relu(L2 @ W3 + b3)  [64x128]  64 WMMAs
//   out = masked column-max over the 64 neighbor rows (0 if ball empty).
// ---------------------------------------------------------------------------
__global__ __launch_bounds__(256) void pointconv_kernel(
    const float* __restrict__ x, const float* __restrict__ pos,
    const float* __restrict__ W1, const float* __restrict__ b1,
    const float* __restrict__ W2, const float* __restrict__ b2,
    const float* __restrict__ W3, const float* __restrict__ b3,
    const float* __restrict__ newPos, const int* __restrict__ nbr,
    const float* __restrict__ nbrD2, float* __restrict__ out) {
  const int gm   = blockIdx.x;        // global centroid id 0..B*M-1
  const int b    = gm >> 11;          // /MM
  const int tid  = threadIdx.x;
  const int lane = tid & 31;
  const int w    = tid >> 5;
  const int ln   = lane & 15;
  const int hi   = lane >> 4;

  __shared__ __align__(32) _Float16 sW1t[64][32];    // W1^T, k padded 6->32
  __shared__ __align__(32) _Float16 sW2t[64][64];    // W2^T
  __shared__ __align__(32) _Float16 sW3t[128][64];   // W3^T
  __shared__ __align__(32) _Float16 sA[64][32];      // features, padded
  __shared__ __align__(32) _Float16 sH[64][64];      // activations (reused)
  __shared__ float sB1[64], sB2[64], sB3[128];
  __shared__ unsigned long long sMask;

  // ---- phase 0a: weights/biases -> LDS (transposed, f16) ----
  if (tid == 0) sMask = 0ull;
  for (int i = tid; i < 64 * 32; i += 256) {
    int n = i >> 5, k = i & 31;
    sW1t[n][k] = (k < 6) ? (_Float16)W1[k * 64 + n] : (_Float16)0.f;
  }
  for (int i = tid; i < 64 * 64; i += 256) {
    int n = i >> 6, k = i & 63;
    sW2t[n][k] = (_Float16)W2[k * 64 + n];
  }
  for (int i = tid; i < 128 * 64; i += 256) {
    int n = i >> 6, k = i & 63;
    sW3t[n][k] = (_Float16)W3[k * 128 + n];
  }
  if (tid < 64) { sB1[tid] = b1[tid]; sB2[tid] = b2[tid]; }
  if (tid < 128) sB3[tid] = b3[tid];
  __syncthreads();   // sMask init + weights visible

  // ---- phase 0b: gather neighbor features [64 x 6] -> sA (f16) ----
  const float qx = newPos[(size_t)gm * 3 + 0];
  const float qy = newPos[(size_t)gm * 3 + 1];
  const float qz = newPos[(size_t)gm * 3 + 2];
  if (tid < 64) {
    const int k = tid;
    const int j = nbr[(size_t)gm * KK + k];
    const float* pj = pos + ((size_t)b * NN + j) * 3;
    const float* xj = x   + ((size_t)b * NN + j) * 3;
    sA[k][0] = (_Float16)xj[0];
    sA[k][1] = (_Float16)xj[1];
    sA[k][2] = (_Float16)xj[2];
    sA[k][3] = (_Float16)(pj[0] - qx);
    sA[k][4] = (_Float16)(pj[1] - qy);
    sA[k][5] = (_Float16)(pj[2] - qz);
#pragma unroll
    for (int c = 6; c < 32; c++) sA[k][c] = (_Float16)0.f;
    if (nbrD2[(size_t)gm * KK + k] <= R2)
      atomicOr(&sMask, 1ull << k);
  }
  __syncthreads();

  // ---- layer 1: [64x32] @ [32x64], 16 tiles, 2 per wave ----
#pragma unroll
  for (int t0 = 0; t0 < 2; t0++) {
    const int t = w * 2 + t0;
    const int mt = t >> 2, nt = t & 3;
    half16 a  = load_afrag(&sA[0][0], 32, mt * 16, 0, lane);
    half16 bf = load_bfrag(&sW1t[0][0], 32, nt * 16, 0, lane);
    float8 acc = F8Z;
    acc = wmma_f16(a, bf, acc);
    const int col = nt * 16 + ln;
    const float bias = sB1[col];
#pragma unroll
    for (int v = 0; v < 8; v++) {
      float val = fmaxf(acc[v] + bias, 0.f);
      sH[mt * 16 + hi * 8 + v][col] = (_Float16)val;
    }
  }
  __syncthreads();

  // ---- layer 2: [64x64] @ [64x64]; compute all reads, barrier, overwrite sH ----
  {
    float8 accs[2];
    int mts[2], nts[2];
#pragma unroll
    for (int t0 = 0; t0 < 2; t0++) {
      const int t = w * 2 + t0;
      mts[t0] = t >> 2; nts[t0] = t & 3;
      float8 acc = F8Z;
#pragma unroll
      for (int kc = 0; kc < 2; kc++) {
        half16 a  = load_afrag(&sH[0][0], 64, mts[t0] * 16, kc * 32, lane);
        half16 bf = load_bfrag(&sW2t[0][0], 64, nts[t0] * 16, kc * 32, lane);
        acc = wmma_f16(a, bf, acc);
      }
      accs[t0] = acc;
    }
    __syncthreads();   // all reads of sH done
#pragma unroll
    for (int t0 = 0; t0 < 2; t0++) {
      const int col = nts[t0] * 16 + ln;
      const float bias = sB2[col];
#pragma unroll
      for (int v = 0; v < 8; v++) {
        float val = fmaxf(accs[t0][v] + bias, 0.f);
        sH[mts[t0] * 16 + hi * 8 + v][col] = (_Float16)val;
      }
    }
    __syncthreads();
  }

  // ---- layer 3: [64x64] @ [64x128]; wave w owns output columns w*16..+15.
  // Masked max over rows folded directly out of the accumulators. ----
  {
    const unsigned long long mb = sMask;
    const int col = w * 16 + ln;
    const float bias = sB3[col];
    float colmax = -FLT_MAX;
#pragma unroll
    for (int mt = 0; mt < 4; mt++) {
      float8 acc = F8Z;
#pragma unroll
      for (int kc = 0; kc < 2; kc++) {
        half16 a  = load_afrag(&sH[0][0], 64, mt * 16, kc * 32, lane);
        half16 bf = load_bfrag(&sW3t[0][0], 64, w * 16, kc * 32, lane);
        acc = wmma_f16(a, bf, acc);
      }
#pragma unroll
      for (int v = 0; v < 8; v++) {
        const int row = mt * 16 + hi * 8 + v;
        float val = fmaxf(acc[v] + bias, 0.f);
        if ((mb >> row) & 1ull) colmax = fmaxf(colmax, val);
      }
    }
    // lanes l and l+16 hold the same column over disjoint row halves
    float o = __shfl_xor(colmax, 16, 32);
    colmax = fmaxf(colmax, o);
    if (lane < 16)
      out[(size_t)gm * 128 + col] = (mb == 0ull) ? 0.f : colmax;
  }
}

// ---------------------------------------------------------------------------
// Host launcher.
// d_out layout (flat, return order): out[8,2048,128] f32 | new_pos[8,2048,3]
// f32 | idx[8,2048] i32 (stored bitwise into the 4-byte slots).
// Workspace: nbr int[8*2048*64] (4MB) | nbrD2 f32[8*2048*64] (4MB).
// ---------------------------------------------------------------------------
extern "C" void kernel_launch(void* const* d_in, const int* in_sizes, int n_in,
                              void* d_out, int out_size, void* d_ws, size_t ws_size,
                              hipStream_t stream) {
  (void)in_sizes; (void)n_in; (void)out_size; (void)ws_size;
  const float* x   = (const float*)d_in[0];
  const float* pos = (const float*)d_in[1];
  const float* W1  = (const float*)d_in[2];
  const float* b1  = (const float*)d_in[3];
  const float* W2  = (const float*)d_in[4];
  const float* b2  = (const float*)d_in[5];
  const float* W3  = (const float*)d_in[6];
  const float* b3  = (const float*)d_in[7];

  float* out    = (float*)d_out;
  float* newPos = out + (size_t)BB * MM * 128;
  int*   outIdx = (int*)(out + (size_t)BB * MM * (128 + 3));

  int*   nbr   = (int*)d_ws;
  float* nbrD2 = (float*)d_ws + (size_t)BB * MM * KK;

  fps_kernel<<<BB, 1024, 0, stream>>>(pos, newPos, outIdx);
  ball_kernel<<<(BB * MM) / 8, 256, 0, stream>>>(pos, newPos, nbr, nbrD2);
  pointconv_kernel<<<BB * MM, 256, 0, stream>>>(x, pos, W1, b1, W2, b2, W3, b3,
                                                newPos, nbr, nbrD2, out);
}
